// SelfAttention_78623671321200
// MI455X (gfx1250) — compile-verified, matrix operand order checked by reading, not run on hardware
//
#include <hip/hip_runtime.h>
#include <hip/hip_bf16.h>
#include <math.h>

typedef __attribute__((ext_vector_type(16))) __bf16 bf16x16;
typedef __attribute__((ext_vector_type(8)))  float  f32x8;

#define BATCH  2
#define LQ     128
#define DMODEL 2048
#define NH     16
#define HD     128
#define SEQ    4096
#define EPS    1e-6f

// ---------------------------------------------------------------------------
// Fragment loaders. CDNA5 16-bit A-matrix 16x32 layout (ISA 7.12.2):
// lane = row (A) / col (B), half = lane>>4.
// elements 0..7  <- k = kbase + half*8 + (0..7)
// elements 8..15 <- k = kbase + 16 + half*8 + (0..7)
// ---------------------------------------------------------------------------
__device__ __forceinline__ bf16x16 frag_from_f32(const float* __restrict__ row,
                                                 int kbase, int half) {
  bf16x16 f;
  const float* p0 = row + kbase + half * 8;
  const float* p1 = p0 + 16;
#pragma unroll
  for (int i = 0; i < 8; ++i) {
    f[i]     = (__bf16)p0[i];
    f[8 + i] = (__bf16)p1[i];
  }
  return f;
}

__device__ __forceinline__ bf16x16 frag_from_bf16(const __bf16* __restrict__ row,
                                                  int kbase, int half) {
  bf16x16 f;
  const __bf16* p0 = row + kbase + half * 8;
  const __bf16* p1 = p0 + 16;
#pragma unroll
  for (int i = 0; i < 8; ++i) {
    f[i]     = p0[i];
    f[8 + i] = p1[i];
  }
  return f;
}

__device__ __forceinline__ f32x8 wmma_bf16(bf16x16 a, bf16x16 b, f32x8 c) {
  return __builtin_amdgcn_wmma_f32_16x16x32_bf16(false, a, false, b,
                                                 (short)0, c, false, false);
}

// ---------------------------------------------------------------------------
// GEMM: C[M,N] = A[M,K] (fp32) @ W[N,K]^T (fp32), bf16 WMMA, fp32 accum/out.
// 1 wave -> one 16x16 C tile. block = 128 threads (4 waves).
// grid.x = N/64, grid.y = M/16.
// ---------------------------------------------------------------------------
__global__ void gemm_bf16_kernel(const float* __restrict__ A,
                                 const float* __restrict__ W,
                                 float* __restrict__ C,
                                 int M, int N, int K) {
  const int wave = threadIdx.x >> 5;
  const int lane = threadIdx.x & 31;
  const int half = lane >> 4;
  const int l16  = lane & 15;
  const int n0   = (blockIdx.x * 4 + wave) * 16;
  const int m0   = blockIdx.y * 16;

  const float* arow = A + (size_t)(m0 + l16) * K;
  const float* wrow = W + (size_t)(n0 + l16) * K;

  f32x8 acc = {};
  for (int kk = 0; kk < K; kk += 32) {
    bf16x16 af = frag_from_f32(arow, kk, half);
    bf16x16 bf = frag_from_f32(wrow, kk, half);
    acc = wmma_bf16(af, bf, acc);
  }
#pragma unroll
  for (int r = 0; r < 8; ++r) {
    const int m = m0 + r + 8 * half;
    C[(size_t)m * N + n0 + l16] = acc[r];
  }
}

// ---------------------------------------------------------------------------
// RMSNorm + RoPE for q,k ; layout conversion for v.
// block = 128 threads, one block per (b, l, h).
// ---------------------------------------------------------------------------
__global__ void prep_qkv_kernel(const float* __restrict__ qp,
                                const float* __restrict__ kp,
                                const float* __restrict__ vp,
                                const float* __restrict__ rope,   // [L*HD]
                                const float* __restrict__ qw,
                                const float* __restrict__ kw,
                                __bf16* __restrict__ qbf,  // [B,NH,LQ,HD]
                                __bf16* __restrict__ kc,   // [B,NH,SEQ,HD]
                                __bf16* __restrict__ vT) { // [B,NH,HD,SEQ]
  const int h = blockIdx.x % NH;
  const int l = (blockIdx.x / NH) % LQ;
  const int b = blockIdx.x / (NH * LQ);
  const int i = threadIdx.x;

  const size_t off = ((size_t)(b * LQ + l) * NH + h) * HD + i;
  const float qv = qp[off];
  const float kv = kp[off];
  const float vv = vp[off];

  __shared__ float sq[128];
  __shared__ float sk[128];
  sq[i] = qv * qv;
  sk[i] = kv * kv;
  __syncthreads();
#pragma unroll
  for (int s = 64; s > 0; s >>= 1) {
    if (i < s) { sq[i] += sq[i + s]; sk[i] += sk[i + s]; }
    __syncthreads();
  }
  const float rq = rsqrtf(sq[0] * (1.0f / HD) + EPS);
  const float rk = rsqrtf(sk[0] * (1.0f / HD) + EPS);
  __syncthreads();

  const float qn = qv * rq * qw[i];
  const float kn = kv * rk * kw[i];

  sq[i] = qn;
  sk[i] = kn;
  __syncthreads();
  const float qrot = (i < 64) ? -sq[i + 64] : sq[i - 64];
  const float krot = (i < 64) ? -sk[i + 64] : sk[i - 64];

  const float fr = rope[l * HD + i];
  const float c = cosf(fr), s = sinf(fr);
  const float qo = qn * c + qrot * s;
  const float ko = kn * c + krot * s;

  qbf[((size_t)(b * NH + h) * LQ + l) * HD + i]              = (__bf16)qo;
  kc [((size_t)(b * NH + h) * SEQ + (SEQ - LQ + l)) * HD + i] = (__bf16)ko;
  vT [((size_t)(b * NH + h) * HD + i) * SEQ + (SEQ - LQ + l)] = (__bf16)vv;
}

// ---------------------------------------------------------------------------
// Convert old cache slots (s < SEQ-LQ) fp32 [B,S,NH,HD] -> bf16 K [B,NH,S,HD]
// and bf16 V^T [B,NH,HD,S]. One block (128 threads) per (b, h, s).
// ---------------------------------------------------------------------------
__global__ void conv_cache_kernel(const float* __restrict__ cache_k,
                                  const float* __restrict__ cache_v,
                                  __bf16* __restrict__ kc,
                                  __bf16* __restrict__ vT) {
  const int SOLD = SEQ - LQ;
  const int s = blockIdx.x % SOLD;
  const int h = (blockIdx.x / SOLD) % NH;
  const int b = blockIdx.x / (SOLD * NH);
  const int i = threadIdx.x;

  const size_t src = (((size_t)b * SEQ + s) * NH + h) * HD + i;
  kc[((size_t)(b * NH + h) * SEQ + s) * HD + i] = (__bf16)cache_k[src];
  vT[((size_t)(b * NH + h) * HD + i) * SEQ + s] = (__bf16)cache_v[src];
}

// ---------------------------------------------------------------------------
// Flash attention: 1 wave per block, one 16-row Q band of one (b,h).
// grid = B*NH*(LQ/16), block = 32.
// ---------------------------------------------------------------------------
__global__ void attn_kernel(const __bf16* __restrict__ qbf,
                            const __bf16* __restrict__ kc,
                            const __bf16* __restrict__ vT,
                            float* __restrict__ out) {  // [B,LQ,DMODEL] fp32
  const int lane = threadIdx.x & 31;
  const int half = lane >> 4;
  const int l16  = lane & 15;
  const int band = blockIdx.x % (LQ / 16);
  const int h    = (blockIdx.x / (LQ / 16)) % NH;
  const int b    = blockIdx.x / ((LQ / 16) * NH);

  const __bf16* qbase = qbf + ((size_t)(b * NH + h) * LQ + band * 16) * HD;
  const __bf16* kbase = kc + (size_t)(b * NH + h) * SEQ * HD;
  const __bf16* vbase = vT + (size_t)(b * NH + h) * HD * SEQ;

  // Preload Q fragments: 4 frags cover d = 128.
  bf16x16 qfrag[4];
  const __bf16* qrow = qbase + (size_t)l16 * HD;
#pragma unroll
  for (int kk = 0; kk < 4; ++kk) qfrag[kk] = frag_from_bf16(qrow, kk * 32, half);

  f32x8 oacc[8];
#pragma unroll
  for (int ct = 0; ct < 8; ++ct) oacc[ct] = (f32x8){};
  float mrow[8], lrow[8];
#pragma unroll
  for (int r = 0; r < 8; ++r) { mrow[r] = -1e30f; lrow[r] = 0.f; }

  __shared__ __bf16 pbuf[16 * 32];
  const float scale = 0.08838834764831845f;  // 1/sqrt(128)

  for (int s0 = 0; s0 < SEQ; s0 += 32) {
    // ---- scores: two 16x16 tiles (cols s0..s0+15 and s0+16..s0+31) ----
    f32x8 t0 = {}, t1 = {};
    const __bf16* krow0 = kbase + (size_t)(s0 + l16) * HD;
    const __bf16* krow1 = kbase + (size_t)(s0 + 16 + l16) * HD;
#pragma unroll
    for (int kk = 0; kk < 4; ++kk)
      t0 = wmma_bf16(qfrag[kk], frag_from_bf16(krow0, kk * 32, half), t0);
#pragma unroll
    for (int kk = 0; kk < 4; ++kk)
      t1 = wmma_bf16(qfrag[kk], frag_from_bf16(krow1, kk * 32, half), t1);

    // ---- online softmax update. C-layout: VGPR r -> row r+8*half,
    //      columns spread over the 16 lanes of each half.
#pragma unroll
    for (int r = 0; r < 8; ++r) {
      float x0 = t0[r] * scale, x1 = t1[r] * scale;
      float mx = fmaxf(x0, x1);
#pragma unroll
      for (int d = 1; d < 16; d <<= 1) mx = fmaxf(mx, __shfl_xor(mx, d, 32));
      const float mnew  = fmaxf(mrow[r], mx);
      const float alpha = __expf(mrow[r] - mnew);
      const float p0 = __expf(x0 - mnew);
      const float p1 = __expf(x1 - mnew);
      float ps = p0 + p1;
#pragma unroll
      for (int d = 1; d < 16; d <<= 1) ps += __shfl_xor(ps, d, 32);
      lrow[r] = lrow[r] * alpha + ps;
      mrow[r] = mnew;
#pragma unroll
      for (int ct = 0; ct < 8; ++ct) oacc[ct][r] *= alpha;
      // P tile -> LDS (row-major 16x32 bf16) for C->A re-layout
      pbuf[(r + 8 * half) * 32 + l16]      = (__bf16)p0;
      pbuf[(r + 8 * half) * 32 + 16 + l16] = (__bf16)p1;
    }
    __syncthreads();

    // ---- P @ V over this 32-wide chunk: out[16 x 128] += P[16x32] @ V[32x128]
    bf16x16 pf = frag_from_bf16(&pbuf[l16 * 32], 0, half);
#pragma unroll
    for (int ct = 0; ct < 8; ++ct) {
      const __bf16* vrow = vbase + (size_t)(ct * 16 + l16) * SEQ;
      oacc[ct] = wmma_bf16(pf, frag_from_bf16(vrow, s0, half), oacc[ct]);
    }
    __syncthreads();
  }

  // ---- normalize and write fp32 attn output [B, LQ, DMODEL]
#pragma unroll
  for (int ct = 0; ct < 8; ++ct) {
#pragma unroll
    for (int r = 0; r < 8; ++r) {
      const int m = r + 8 * half;
      const float v = oacc[ct][r] / lrow[r];
      out[((size_t)b * LQ + band * 16 + m) * DMODEL + h * HD + ct * 16 + l16] = v;
    }
  }
}

// ---------------------------------------------------------------------------
extern "C" void kernel_launch(void* const* d_in, const int* in_sizes, int n_in,
                              void* d_out, int out_size, void* d_ws, size_t ws_size,
                              hipStream_t stream) {
  const float* x       = (const float*)d_in[0];
  const float* rope    = (const float*)d_in[1];
  const float* cache_k = (const float*)d_in[2];
  const float* cache_v = (const float*)d_in[3];
  const float* wq      = (const float*)d_in[4];
  const float* wk      = (const float*)d_in[5];
  const float* wv      = (const float*)d_in[6];
  const float* wo      = (const float*)d_in[7];
  const float* qnw     = (const float*)d_in[8];
  const float* knw     = (const float*)d_in[9];

  char* p = (char*)d_ws;
  const size_t projElems = (size_t)BATCH * LQ * DMODEL;
  float* qproj = (float*)p; p += projElems * sizeof(float);
  float* kproj = (float*)p; p += projElems * sizeof(float);
  float* vproj = (float*)p; p += projElems * sizeof(float);
  float* attnb = (float*)p; p += projElems * sizeof(float);
  __bf16* qbf = (__bf16*)p; p += (size_t)BATCH * NH * LQ * HD * sizeof(__bf16);
  __bf16* kc  = (__bf16*)p; p += (size_t)BATCH * NH * SEQ * HD * sizeof(__bf16);
  __bf16* vT  = (__bf16*)p; p += (size_t)BATCH * NH * HD * SEQ * sizeof(__bf16);

  const dim3 ggrid(DMODEL / 64, (BATCH * LQ) / 16);
  gemm_bf16_kernel<<<ggrid, 128, 0, stream>>>(x, wq, qproj, BATCH * LQ, DMODEL, DMODEL);
  gemm_bf16_kernel<<<ggrid, 128, 0, stream>>>(x, wk, kproj, BATCH * LQ, DMODEL, DMODEL);
  gemm_bf16_kernel<<<ggrid, 128, 0, stream>>>(x, wv, vproj, BATCH * LQ, DMODEL, DMODEL);

  prep_qkv_kernel<<<BATCH * LQ * NH, 128, 0, stream>>>(qproj, kproj, vproj, rope,
                                                       qnw, knw, qbf, kc, vT);
  conv_cache_kernel<<<BATCH * NH * (SEQ - LQ), 128, 0, stream>>>(cache_k, cache_v, kc, vT);

  attn_kernel<<<BATCH * NH * (LQ / 16), 32, 0, stream>>>(qbf, kc, vT, attnb);

  gemm_bf16_kernel<<<ggrid, 128, 0, stream>>>(attnb, wo, (float*)d_out,
                                              BATCH * LQ, DMODEL, DMODEL);
}
